// SimpleVanillaRFRNN_89756226552406
// MI455X (gfx1250) — matrix operand order repaired and checked by reading, not compiled, not run on hardware
//
#include <hip/hip_runtime.h>
#include <hip/hip_bf16.h>

// ---------------- types for WMMA fragments (wave32, gfx1250) ----------------
typedef __bf16 bf16;
typedef __attribute__((ext_vector_type(16))) __bf16 v16bf;
typedef __attribute__((ext_vector_type(8)))  __bf16 v8bf;
typedef __attribute__((ext_vector_type(4)))  __bf16 v4bf;
typedef __attribute__((ext_vector_type(8)))  float  v8f;
typedef __attribute__((ext_vector_type(4)))  int    v4i;

// ---------------- problem constants ----------------
constexpr int T_  = 500;
constexpr int B_  = 256;
constexpr int I_  = 256;
constexpr int H_  = 1024;
constexpr int O_  = 128;
constexpr int IH_ = 1280;   // I + H
#define DT_C 0.01f

// LDS strip leading dims: +8 bf16 pad (16B) so 16-row ds_load_b128 fragment
// reads rotate across banks (row stride 516 dwords -> 4-bank shift per row).
constexpr int XP = I_ + 8;   // 264
constexpr int ZP = H_ + 8;   // 1032
constexpr int SMEM_ELTS = 16 * XP + 16 * ZP;   // 41,472 bytes

// ---- CDNA5 async global->LDS copy (ASYNCcnt path), with fallback ----
#if defined(__has_builtin)
#if __has_builtin(__builtin_amdgcn_global_load_async_to_lds_b128) && \
    __has_builtin(__builtin_amdgcn_s_wait_asynccnt)
#define USE_ASYNC_COPY 1
#endif
#endif

#ifdef USE_ASYNC_COPY
// builtin signature (from clang diagnostic): (int4 AS1*, int4 AS3*, Imm, Imm)
typedef __attribute__((address_space(1))) v4i as1_v4i;
typedef __attribute__((address_space(3))) v4i as3_v4i;
#endif

__device__ __forceinline__ void copy16_g2l(bf16* lds_dst, const bf16* gsrc) {
#ifdef USE_ASYNC_COPY
  __builtin_amdgcn_global_load_async_to_lds_b128((as1_v4i*)gsrc,
                                                 (as3_v4i*)lds_dst, 0, 0);
#else
  *(v8bf*)lds_dst = *(const v8bf*)gsrc;
#endif
}

__device__ __forceinline__ void copy_fence_and_barrier() {
#ifdef USE_ASYNC_COPY
  __builtin_amdgcn_s_wait_asynccnt(0);
#endif
  __syncthreads();
}

// ============================================================================
// Fragment loaders per CDNA5 ISA §7.12.2 VGPR layouts (wave32).
// A: 16x32 bf16, row M = lane&15; lanes 0-15 hold K {k0+0..7, k0+16..23},
//    lanes 16-31 hold K {k0+8..15, k0+24..31}. Two 16B loads per lane.
// B: 32x16 bf16: column N = lane&15; lanes 0-15 hold K {k0+0..7, k0+16..23},
//    lanes 16-31 hold K {k0+8..15, k0+24..31}. Two 16B loads per lane.
// ============================================================================
__device__ __forceinline__ v16bf load_a_frag_lds(const bf16* __restrict__ strip,
                                                 int lda, int k0, int lane) {
  const bf16* p = strip + (lane & 15) * lda + k0 + ((lane & 16) ? 8 : 0);
  v8bf lo = *(const v8bf*)(p);        // ds_load_b128
  v8bf hi = *(const v8bf*)(p + 16);
  v16bf a;
#pragma unroll
  for (int i = 0; i < 8; ++i) { a[i] = lo[i]; a[8 + i] = hi[i]; }
  return a;
}

__device__ __forceinline__ v16bf load_b_frag(const bf16* __restrict__ base,
                                             int ldb, int col, int k0, int lane) {
  const bf16* p = base + (long)col * ldb + k0 + ((lane & 16) ? 8 : 0);
  v8bf lo = *(const v8bf*)(p);
  v8bf hi = *(const v8bf*)(p + 16);
  v16bf b;
#pragma unroll
  for (int i = 0; i < 8; ++i) { b[i] = lo[i]; b[8 + i] = hi[i]; }
  return b;
}

// ============================================================================
// One-time f32 -> bf16 conversion (vectorized 4-wide)
// ============================================================================
__global__ void k_f32_to_bf16_v4(const float* __restrict__ in,
                                 bf16* __restrict__ out, long n4) {
  long i      = (long)blockIdx.x * blockDim.x + threadIdx.x;
  long stride = (long)gridDim.x * blockDim.x;
  for (; i < n4; i += stride) {
    float4 f = ((const float4*)in)[i];
    v4bf o;
    o[0] = (bf16)f.x; o[1] = (bf16)f.y; o[2] = (bf16)f.z; o[3] = (bf16)f.w;
    ((v4bf*)out)[i] = o;
  }
}

// w = |omega|, b = -|b_offset|, alpha = exp(-DT/|tau|)
__global__ void k_params(const float* __restrict__ omega,
                         const float* __restrict__ boff,
                         const float* __restrict__ tau,
                         float* __restrict__ wvec, float* __restrict__ bvec,
                         float* __restrict__ alpha) {
  int i = blockIdx.x * blockDim.x + threadIdx.x;
  if (i < H_) { wvec[i] = fabsf(omega[i]); bvec[i] = -fabsf(boff[i]); }
  if (i < O_) { alpha[i] = __expf(-DT_C / fabsf(tau[i])); }
}

// zero all recurrent state (fresh every kernel_launch call -> deterministic)
__global__ void k_init(float* __restrict__ u, float* __restrict__ v,
                       float* __restrict__ ou, bf16* __restrict__ zb0,
                       float* __restrict__ num_spikes) {
  int i = blockIdx.x * blockDim.x + threadIdx.x;
  if (i < B_ * H_) { u[i] = 0.f; v[i] = 0.f; zb0[i] = (bf16)0.f; }
  if (i < B_ * O_) { ou[i] = 0.f; }
  if (i == 0) { *num_spikes = 0.f; }
}

// ============================================================================
// Fused per-step kernel (launched for t = 0..T):
//   blocks 0..127  : HIDDEN for step t      (active when t <  T)
//     cur = x_t @ Wx^T + z_{t-1} @ Wz^T ; RF update; z_t = (u' > 1)
//     1024 waves, one 16x16 tile each; block = 1 tm x 8 tn -> the 16-row A
//     strip (x rows + z rows) is shared by all 8 waves and staged in LDS once.
//   blocks 128..143: READOUT for step t-1   (active when t >= 1)
//     ou = a*ou + (1-a)*(z_{t-1} @ Wout^T) ; outputs[t-1] = ou
//     128 waves; block = 1 tm x 8 tn; same LDS staging of the z strip.
// Both roles read z buffer (t&1); hidden writes the other buffer -> no hazard.
// ============================================================================
__global__ __launch_bounds__(256) void k_step(
    int t,
    const bf16* __restrict__ xb,    // [T,B,I]
    const bf16* __restrict__ wh,    // [H,IH]
    const bf16* __restrict__ wout,  // [O,H]
    const bf16* __restrict__ z_in,  // [B,H]  z_{t-1}
    bf16* __restrict__ z_out,       // [B,H]  z_t
    float* __restrict__ u, float* __restrict__ v,
    const float* __restrict__ wvec, const float* __restrict__ bvec,
    const float* __restrict__ alpha, float* __restrict__ ou,
    float* __restrict__ outputs,    // d_out: [T,B,O]
    float* __restrict__ num_spikes) {
  __shared__ __align__(16) bf16 smem[SMEM_ELTS];
  bf16* xs = smem;             // 16 x XP  (hidden only)
  bf16* zs = smem + 16 * XP;   // 16 x ZP

  const int tid  = threadIdx.x;
  const int lane = tid & 31;
  const int wv   = tid >> 5;

  if (blockIdx.x < 128) {
    // ---------------- HIDDEN role ----------------
    if (t >= T_) return;
    const int tm = blockIdx.x >> 3;              // 0..15 batch tile
    const int tn = (blockIdx.x & 7) * 8 + wv;    // 0..63 hidden tile
    const bf16* xt = xb + (long)t * B_ * I_;

    // stage A strips: x rows (16 x 256) and z rows (16 x 1024), 16B chunks
    {
      const bf16* xsrc = xt + (long)tm * 16 * I_;
#pragma unroll
      for (int c = tid; c < 512; c += 256) {
        int r = c >> 5, o = (c & 31) * 8;
        copy16_g2l(xs + r * XP + o, xsrc + r * I_ + o);
      }
      const bf16* zsrc = z_in + (long)tm * 16 * H_;
#pragma unroll
      for (int c = tid; c < 2048; c += 256) {
        int r = c >> 7, o = (c & 127) * 8;
        copy16_g2l(zs + r * ZP + o, zsrc + r * H_ + o);
      }
      copy_fence_and_barrier();
    }

    const int bcol = tn * 16 + (lane & 15);
    v8f acc = {};
#pragma unroll 4
    for (int k0 = 0; k0 < I_; k0 += 32) {
      v16bf a = load_a_frag_lds(xs, XP, k0, lane);
      v16bf b = load_b_frag(wh, IH_, bcol, k0, lane);
      acc = __builtin_amdgcn_wmma_f32_16x16x32_bf16(false, a, false, b,
                                                    (short)0, acc, false, false);
    }
#pragma unroll 4
    for (int k0 = 0; k0 < H_; k0 += 32) {
      v16bf a = load_a_frag_lds(zs, ZP, k0, lane);
      v16bf b = load_b_frag(wh, IH_, bcol, I_ + k0, lane);
      acc = __builtin_amdgcn_wmma_f32_16x16x32_bf16(false, a, false, b,
                                                    (short)0, acc, false, false);
    }

    // epilogue: C/D layout -> lanes 0-15: M=r; lanes 16-31: M=r+8; N=lane&15
    const float w_ = wvec[bcol];
    const float b_ = bvec[bcol];
    float spikes = 0.f;
#pragma unroll
    for (int r = 0; r < 8; ++r) {
      const int  row = tm * 16 + r + ((lane & 16) ? 8 : 0);
      const long idx = (long)row * H_ + bcol;
      const float uo = u[idx], vo = v[idx];
      const float un = uo + DT_C * (b_ * uo - w_ * vo + acc[r]);
      const float vn = vo + DT_C * (w_ * un + b_ * vo);
      const float z  = (un > 1.0f) ? 1.0f : 0.0f;   // spike(u - THETA)
      u[idx] = un;
      v[idx] = vn;
      z_out[idx] = (bf16)z;                          // exact in bf16
      spikes += z;
    }
#pragma unroll
    for (int off = 16; off > 0; off >>= 1)
      spikes += __shfl_down(spikes, off, 32);
    if (lane == 0) atomicAdd(num_spikes, spikes);
  } else {
    // ---------------- READOUT role (step t-1) ----------------
    if (t < 1) return;
    const int ts = t - 1;
    const int tm = blockIdx.x - 128;   // 0..15 batch tile
    const int tn = wv;                 // 0..7  output tile

    {
      const bf16* zsrc = z_in + (long)tm * 16 * H_;
#pragma unroll
      for (int c = tid; c < 2048; c += 256) {
        int r = c >> 7, o = (c & 127) * 8;
        copy16_g2l(zs + r * ZP + o, zsrc + r * H_ + o);
      }
      copy_fence_and_barrier();
    }

    const int bcol = tn * 16 + (lane & 15);
    v8f acc = {};
#pragma unroll 4
    for (int k0 = 0; k0 < H_; k0 += 32) {
      v16bf a = load_a_frag_lds(zs, ZP, k0, lane);
      v16bf b = load_b_frag(wout, H_, bcol, k0, lane);
      acc = __builtin_amdgcn_wmma_f32_16x16x32_bf16(false, a, false, b,
                                                    (short)0, acc, false, false);
    }

    const float al = alpha[bcol];
    float* outp = outputs + (long)ts * B_ * O_;
#pragma unroll
    for (int r = 0; r < 8; ++r) {
      const int  row = tm * 16 + r + ((lane & 16) ? 8 : 0);
      const long idx = (long)row * O_ + bcol;
      const float on = al * ou[idx] + (1.0f - al) * acc[r];
      ou[idx]   = on;
      outp[idx] = on;
    }
  }
}

// write final state outputs: z (bf16 -> f32, exact), u, ou
__global__ void k_finalize(const bf16* __restrict__ zfin,
                           const float* __restrict__ u,
                           const float* __restrict__ ou,
                           float* __restrict__ out_z,
                           float* __restrict__ out_u,
                           float* __restrict__ out_ou) {
  int i = blockIdx.x * blockDim.x + threadIdx.x;
  if (i < B_ * H_) { out_z[i] = (float)zfin[i]; out_u[i] = u[i]; }
  if (i < B_ * O_) { out_ou[i] = ou[i]; }
}

// ============================================================================
// Host launcher
// ============================================================================
extern "C" void kernel_launch(void* const* d_in, const int* in_sizes, int n_in,
                              void* d_out, int out_size, void* d_ws, size_t ws_size,
                              hipStream_t stream) {
  const float* x     = (const float*)d_in[0];  // [T,B,I]
  const float* Wh    = (const float*)d_in[1];  // [H,IH]
  const float* omega = (const float*)d_in[2];  // [H]
  const float* boff  = (const float*)d_in[3];  // [H]
  const float* Wout  = (const float*)d_in[4];  // [O,H]
  const float* tau   = (const float*)d_in[5];  // [O]
  float* out = (float*)d_out;

  // ---- workspace layout (256B aligned slots) ----
  char* ws = (char*)d_ws;
  auto au = [](size_t v) { return (v + 255) & ~(size_t)255; };
  size_t off = 0;
  bf16* xb    = (bf16*)(ws + off); off = au(off + (size_t)T_ * B_ * I_ * 2);
  bf16* whb   = (bf16*)(ws + off); off = au(off + (size_t)H_ * IH_ * 2);
  bf16* woutb = (bf16*)(ws + off); off = au(off + (size_t)O_ * H_ * 2);
  bf16* zb0   = (bf16*)(ws + off); off = au(off + (size_t)B_ * H_ * 2);
  bf16* zb1   = (bf16*)(ws + off); off = au(off + (size_t)B_ * H_ * 2);
  float* u    = (float*)(ws + off); off = au(off + (size_t)B_ * H_ * 4);
  float* v    = (float*)(ws + off); off = au(off + (size_t)B_ * H_ * 4);
  float* ou   = (float*)(ws + off); off = au(off + (size_t)B_ * O_ * 4);
  float* wvec = (float*)(ws + off); off = au(off + (size_t)H_ * 4);
  float* bvec = (float*)(ws + off); off = au(off + (size_t)H_ * 4);
  float* alph = (float*)(ws + off); off = au(off + (size_t)O_ * 4);

  // ---- output layout: outputs | z | u | ou | num_spikes ----
  float* out_outputs = out;
  float* out_z       = out + (size_t)T_ * B_ * O_;
  float* out_u       = out_z + (size_t)B_ * H_;
  float* out_ou      = out_u + (size_t)B_ * H_;
  float* out_ns      = out_ou + (size_t)B_ * O_;

  // 1) one-time bf16 conversion
  k_f32_to_bf16_v4<<<4096, 256, 0, stream>>>(x, xb, (long)T_ * B_ * I_ / 4);
  k_f32_to_bf16_v4<<<640, 256, 0, stream>>>(Wh, whb, (long)H_ * IH_ / 4);
  k_f32_to_bf16_v4<<<128, 256, 0, stream>>>(Wout, woutb, (long)O_ * H_ / 4);
  k_params<<<4, 256, 0, stream>>>(omega, boff, tau, wvec, bvec, alph);

  // 2) zero recurrent state + spike counter
  k_init<<<(B_ * H_ + 255) / 256, 256, 0, stream>>>(u, v, ou, zb0, out_ns);

  // 3) fused scan: launch t does hidden(t) + readout(t-1); t == T is the
  //    trailing readout-only launch. Both roles read zbuf[t&1]; hidden(t)
  //    writes zbuf[(t+1)&1].
  for (int t = 0; t <= T_; ++t) {
    bf16* z_in  = (t & 1) ? zb1 : zb0;
    bf16* z_out = (t & 1) ? zb0 : zb1;
    k_step<<<144, 256, 0, stream>>>(t, xb, whb, woutb, z_in, z_out, u, v,
                                    wvec, bvec, alph, ou, out_outputs, out_ns);
  }

  // 4) final state: hidden(T-1) wrote zbuf[T&1] == zb0 (T even)
  bf16* zfin = (T_ & 1) ? zb1 : zb0;
  k_finalize<<<(B_ * H_ + 255) / 256, 256, 0, stream>>>(zfin, u, ou,
                                                        out_z, out_u, out_ou);
}
